// VSSBlock_45243185496314
// MI455X (gfx1250) — matrix operand order repaired and checked by reading, not compiled, not run on hardware
//
#include <hip/hip_runtime.h>
#include <hip/hip_bf16.h>

typedef __attribute__((ext_vector_type(16))) _Float16 v16h;
typedef __attribute__((ext_vector_type(8)))  _Float16 v8h;
typedef __attribute__((ext_vector_type(8)))  float    v8f;

#define BN 2
#define HN 48
#define WN 48
#define LN (HN*WN)      // 2304
#define DM 64
#define DI 128
#define NS 16
#define RT 4
#define KD 4
#define NP (BN*LN)      // 4608
#define CW 48           // padded x_dbl row width (36 valid -> 48)
#define CPAD 48         // padded x_proj_w column count

// A-fragment for v_wmma_f32_16x16x32_f16: per ISA layout this is two
// contiguous 8-half (16B) chunks at K = kk+8*half and K = kk+16+8*half.
__device__ __forceinline__ v16h load_a_frag(const _Float16* __restrict__ arow,
                                            int kk, int half) {
    v8h a0 = *(const v8h*)(arow + kk + 8*half);
    v8h a1 = *(const v8h*)(arow + kk + 16 + 8*half);
    return __builtin_shufflevector(a0, a1, 0,1,2,3,4,5,6,7,8,9,10,11,12,13,14,15);
}
// B-fragment: two contiguous 8-half (16B) chunks at K = kk+8*half and K = kk+16+8*half (lane = N).
__device__ __forceinline__ v16h load_b_frag(const _Float16* __restrict__ brow,
                                            int kk, int half) {
    v8h b0 = *(const v8h*)(brow + kk + 8*half);
    v8h b1 = *(const v8h*)(brow + kk + 16 + 8*half);
    return __builtin_shufflevector(b0, b1, 0,1,2,3,4,5,6,7,8,9,10,11,12,13,14,15);
}

// ---------------- pack all GEMM weights to f16 (x_proj_w padded 36->48) ----------------
__global__ void k_pack(const float* __restrict__ w_in, const float* __restrict__ xpw,
                       const float* __restrict__ wout,
                       _Float16* __restrict__ winh, _Float16* __restrict__ xpwh,
                       _Float16* __restrict__ wouth) {
    int i = blockIdx.x * blockDim.x + threadIdx.x;
    if (i < 256*DM) {
        winh[i] = (_Float16)w_in[i];
    } else if (i < 256*DM + KD*CPAD*DI) {
        int j = i - 256*DM;
        int k = j / (CPAD*DI), rem = j % (CPAD*DI);
        int c = rem / DI, d = rem % DI;
        xpwh[j] = (_Float16)((c < 36) ? xpw[(size_t)(k*36 + c)*DI + d] : 0.0f);
    } else {
        int j = i - 256*DM - KD*CPAD*DI;
        wouth[j] = (_Float16)wout[j];
    }
}

// ---------------- LayerNorm (d_model=64) -> f16, 8 positions per block ----------------
__global__ void k_ln(const float* __restrict__ x, const float* __restrict__ g,
                     const float* __restrict__ bt, _Float16* __restrict__ xnh) {
    int p = blockIdx.x * 8 + (threadIdx.x >> 5);
    int lane = threadIdx.x & 31;
    float v0 = x[(size_t)p*DM + lane];
    float v1 = x[(size_t)p*DM + lane + 32];
    float s = v0 + v1, q = v0*v0 + v1*v1;
    #pragma unroll
    for (int m = 16; m > 0; m >>= 1) {
        s += __shfl_xor(s, m, 32);
        q += __shfl_xor(q, m, 32);
    }
    float mean = s * (1.0f/64.0f);
    float var  = q * (1.0f/64.0f) - mean*mean;
    float inv  = rsqrtf(var + 1e-6f);
    xnh[(size_t)p*DM + lane]      = (_Float16)((v0 - mean)*inv*g[lane]    + bt[lane]);
    xnh[(size_t)p*DM + lane + 32] = (_Float16)((v1 - mean)*inv*g[lane+32] + bt[lane+32]);
}

// ---------------- in_proj GEMM: [4608x64] x [64x256] via WMMA ----------------
__global__ void k_gemm_inproj(const _Float16* __restrict__ xnh,
                              const _Float16* __restrict__ winh,  // [256][64] f16
                              float* __restrict__ Xi,             // [b][128][L]
                              float* __restrict__ Z) {            // [p][128]
    int tile = blockIdx.x * 4 + (threadIdx.x >> 5);
    int nt = tile & 15;          // 256/16 n-tiles
    int mt = tile >> 4;          // 288 m-tiles
    int lane = threadIdx.x & 31, row = lane & 15, half = lane >> 4;
    const _Float16* arow = xnh  + (size_t)(mt*16 + row)*DM;
    const _Float16* brow = winh + (size_t)(nt*16 + row)*DM;
    v8f acc = {};
    #pragma unroll
    for (int kk = 0; kk < DM; kk += 32) {
        v16h a  = load_a_frag(arow, kk, half);
        v16h bf = load_b_frag(brow, kk, half);
        acc = __builtin_amdgcn_wmma_f32_16x16x32_f16(false, a, false, bf,
                                                     (short)0, acc, false, false);
    }
    int n  = nt*16 + row;
    int m0 = mt*16;
    int b_ = m0 / LN;            // tile never crosses batch (2304 % 16 == 0)
    int l0 = m0 % LN;
    if (nt < 8) {                // uniform per tile: xi half -> NCHW, contiguous in l
        float* xp = Xi + (size_t)(b_*DI + n)*LN + l0 + 8*half;
        #pragma unroll
        for (int r = 0; r < 8; ++r) xp[r] = acc[r];
    } else {                     // z half
        float* zp = Z + (size_t)(m0 + 8*half)*DI + (n - DI);
        #pragma unroll
        for (int r = 0; r < 8; ++r) zp[(size_t)r*DI] = acc[r];
    }
}

// ---------------- depthwise 3x3 conv + bias + SiLU ----------------
__global__ void k_conv(const float* __restrict__ Xi, const float* __restrict__ cw,
                       const float* __restrict__ cb,
                       float* __restrict__ xcf,        // f32 [b][d][L]
                       _Float16* __restrict__ xch) {   // f16 [b][L][d]
    int idx = blockIdx.x * blockDim.x + threadIdx.x;   // b*DI*LN total
    int b_ = idx / (DI*LN);
    int rem = idx % (DI*LN);
    int d = rem / LN, l = rem % LN;
    int h = l / WN, w = l % WN;
    const float* in = Xi + (size_t)(b_*DI + d)*LN;
    const float* wt = cw + d*9;
    float s = cb[d];
    #pragma unroll
    for (int i = 0; i < 3; ++i) {
        int hh = h + i - 1;
        if (hh < 0 || hh >= HN) continue;
        #pragma unroll
        for (int j = 0; j < 3; ++j) {
            int ww = w + j - 1;
            if (ww < 0 || ww >= WN) continue;
            s += in[hh*WN + ww] * wt[i*3 + j];
        }
    }
    float sil = s / (1.0f + expf(-s));
    xcf[idx] = sil;
    xch[(size_t)(b_*LN + l)*DI + d] = (_Float16)sil;
}

// Direction index maps (all involutions)
__device__ __forceinline__ int map_dir(int k, int l) {
    if (k == 0) return l;
    if (k == 1) { int h = l / WN, w = l % WN; return w * WN + h; }
    if (k == 2) return LN - 1 - l;
    int l2 = LN - 1 - l; int h = l2 / WN, w = l2 % WN; return w * WN + h;
}

// ---------------- x_proj GEMM per (b,k): [Lx128] x [128x48pad] via WMMA ----------------
__global__ void k_gemm_xproj(const _Float16* __restrict__ xch,   // [b][l][128]
                             const _Float16* __restrict__ xpwh,  // [4][48][128] f16
                             float* __restrict__ xdbl) {         // [b][4][l][48]
    int tile = blockIdx.x * 4 + (threadIdx.x >> 5);
    int nt = tile % 3;
    int mt = (tile / 3) % (LN/16);
    int k  = (tile / (3*(LN/16))) % KD;
    int b_ =  tile / (3*(LN/16)*KD);
    int lane = threadIdx.x & 31, row = lane & 15, half = lane >> 4;
    int lp = map_dir(k, mt*16 + row);            // xs[b,k,:,l] = xc[b,:,map_k(l)]
    const _Float16* arow = xch  + (size_t)(b_*LN + lp)*DI;
    const _Float16* brow = xpwh + (size_t)(k*CPAD + nt*16 + row)*DI;
    v8f acc = {};
    #pragma unroll
    for (int kk = 0; kk < DI; kk += 32) {
        v16h a  = load_a_frag(arow, kk, half);
        v16h bf = load_b_frag(brow, kk, half);
        acc = __builtin_amdgcn_wmma_f32_16x16x32_f16(false, a, false, bf,
                                                     (short)0, acc, false, false);
    }
    int c = nt*16 + row;
    float* op = xdbl + ((size_t)((b_*KD + k)*LN + mt*16 + 8*half))*CW + c;
    #pragma unroll
    for (int r = 0; r < 8; ++r) op[(size_t)r*CW] = acc[r];
}

// ---------------- selective scan: 16 lanes per (b,k,d), one state elem per lane ----------------
__global__ void k_scan(const float* __restrict__ xdbl, const float* __restrict__ xcf,
                       const float* __restrict__ dtw, const float* __restrict__ dtb,
                       const float* __restrict__ Alogs, const float* __restrict__ Ds,
                       float* __restrict__ ybuf) {     // [b][k][d][l]
    int tid = threadIdx.x;
    int g = blockIdx.x * (blockDim.x >> 4) + (tid >> 4);   // channel id 0..1023
    int n = tid & 15;                                      // state index
    int b_ = g >> 9;
    int k  = (g >> 7) & 3;      // uniform per block (16 groups never cross k)
    int d  = g & 127;
    int kd = k*DI + d;
    float An  = -expf(Alogs[(size_t)kd*NS + n]);
    float Dv  = Ds[kd];
    float w0 = dtw[kd*RT+0], w1 = dtw[kd*RT+1], w2 = dtw[kd*RT+2], w3 = dtw[kd*RT+3];
    float bias = dtb[kd];
    const float* xrow = xcf  + (size_t)(b_*DI + d)*LN;
    const float* xd   = xdbl + (size_t)((b_*KD + k)*LN)*CW;
    float* yout = ybuf + (size_t)((b_*KD + k)*DI + d)*LN;
    float h = 0.0f;
    int hh = 0, ww = 0;                                    // incremental (h,w) of l
    for (int l = 0; l < LN; ++l) {
        const float* row = xd + (size_t)l*CW;
        if (l + 1 < LN) __builtin_prefetch(row + CW, 0, 0);
        int lp;
        switch (k) {                                       // uniform per block
            case 0:  lp = l; break;
            case 1:  lp = ww*WN + hh; break;
            case 2:  lp = LN - 1 - l; break;
            default: lp = (WN-1-ww)*WN + (HN-1-hh); break;
        }
        float pre = row[0]*w0 + row[1]*w1 + row[2]*w2 + row[3]*w3 + bias;
        float delta = (pre > 20.0f) ? pre : log1pf(expf(pre));   // softplus
        float xv = xrow[lp];
        float a = expf(delta * An);
        h = h * a + delta * xv * row[4 + n];     // B_n
        float t = h * row[20 + n];               // C_n
        t += __shfl_xor(t, 1, 16);
        t += __shfl_xor(t, 2, 16);
        t += __shfl_xor(t, 4, 16);
        t += __shfl_xor(t, 8, 16);
        if (n == 0) yout[l] = t + Dv * xv;
        if (++ww == WN) { ww = 0; ++hh; }
    }
}

// ---------------- combine 4 directions + out_norm + SiLU(z) gate -> f16 ----------------
__global__ void k_combine(const float* __restrict__ ybuf, const float* __restrict__ Z,
                          const float* __restrict__ ong, const float* __restrict__ onb,
                          _Float16* __restrict__ yh16) {
    __shared__ float red[128];
    __shared__ float red2[128];
    __shared__ float sm[2];
    int p = blockIdx.x;
    int b_ = p / LN, l = p % LN;
    int d = threadIdx.x;
    float s = 0.0f;
    #pragma unroll
    for (int k = 0; k < KD; ++k) {
        int lp = map_dir(k, l);                  // involution: un-map == map
        s += ybuf[(size_t)((b_*KD + k)*DI + d)*LN + lp];
    }
    red[d] = s; red2[d] = s*s;
    __syncthreads();
    for (int st = 64; st > 0; st >>= 1) {
        if (d < st) { red[d] += red[d+st]; red2[d] += red2[d+st]; }
        __syncthreads();
    }
    if (d == 0) {
        float mean = red[0] * (1.0f/128.0f);
        float var  = red2[0] * (1.0f/128.0f) - mean*mean;
        sm[0] = mean; sm[1] = rsqrtf(var + 1e-5f);
    }
    __syncthreads();
    float yn = (s - sm[0]) * sm[1] * ong[d] + onb[d];
    float zv = Z[(size_t)p*DI + d];
    float gate = zv / (1.0f + expf(-zv));        // SiLU
    yh16[(size_t)p*DI + d] = (_Float16)(yn * gate);
}

// ---------------- out_proj GEMM + residual: [4608x128] x [128x64] via WMMA ----------------
__global__ void k_gemm_out(const _Float16* __restrict__ yh16,
                           const _Float16* __restrict__ wouth,  // [64][128] f16
                           const float* __restrict__ x,
                           float* __restrict__ out) {
    int tile = blockIdx.x * 4 + (threadIdx.x >> 5);
    int nt = tile & 3;           // 64/16 n-tiles
    int mt = tile >> 2;          // 288 m-tiles
    int lane = threadIdx.x & 31, row = lane & 15, half = lane >> 4;
    int n = nt*16 + row;
    const _Float16* arow = yh16  + (size_t)(mt*16 + row)*DI;
    const _Float16* brow = wouth + (size_t)n*DI;
    v8f acc = {};
    #pragma unroll
    for (int kk = 0; kk < DI; kk += 32) {
        v16h a  = load_a_frag(arow, kk, half);
        v16h bf = load_b_frag(brow, kk, half);
        acc = __builtin_amdgcn_wmma_f32_16x16x32_f16(false, a, false, bf,
                                                     (short)0, acc, false, false);
    }
    const float* xp = x + (size_t)(mt*16 + 8*half)*DM + n;
    float* op = out + (size_t)(mt*16 + 8*half)*DM + n;
    #pragma unroll
    for (int r = 0; r < 8; ++r) op[(size_t)r*DM] = xp[(size_t)r*DM] + acc[r];
}

extern "C" void kernel_launch(void* const* d_in, const int* in_sizes, int n_in,
                              void* d_out, int out_size, void* d_ws, size_t ws_size,
                              hipStream_t stream) {
    const float* x         = (const float*)d_in[0];
    const float* ln1_g     = (const float*)d_in[1];
    const float* ln1_b     = (const float*)d_in[2];
    const float* in_proj_w = (const float*)d_in[3];
    const float* conv_w    = (const float*)d_in[4];
    const float* conv_b    = (const float*)d_in[5];
    const float* x_proj_w  = (const float*)d_in[6];
    const float* dt_proj_w = (const float*)d_in[7];
    const float* dt_proj_b = (const float*)d_in[8];
    const float* A_logs    = (const float*)d_in[9];
    const float* Ds        = (const float*)d_in[10];
    const float* out_ng    = (const float*)d_in[11];
    const float* out_nb    = (const float*)d_in[12];
    const float* out_pw    = (const float*)d_in[13];
    float* out = (float*)d_out;

    // workspace carve-out (~25 MB total)
    char* wsp = (char*)d_ws;
    auto carve = [&](size_t bytes) {
        char* r = wsp; wsp += (bytes + 255) & ~(size_t)255; return r;
    };
    _Float16* xnh   = (_Float16*)carve((size_t)NP*DM*sizeof(_Float16));
    float*    Xi    = (float*)   carve((size_t)BN*DI*LN*sizeof(float));
    float*    Z     = (float*)   carve((size_t)NP*DI*sizeof(float));
    float*    xcf   = (float*)   carve((size_t)BN*DI*LN*sizeof(float));
    _Float16* xch   = (_Float16*)carve((size_t)NP*DI*sizeof(_Float16));
    float*    xdbl  = (float*)   carve((size_t)BN*KD*LN*CW*sizeof(float));
    float*    ybuf  = (float*)   carve((size_t)BN*KD*DI*LN*sizeof(float));
    _Float16* yh16  = (_Float16*)carve((size_t)NP*DI*sizeof(_Float16));
    _Float16* winh  = (_Float16*)carve((size_t)256*DM*sizeof(_Float16));
    _Float16* xpwh  = (_Float16*)carve((size_t)KD*CPAD*DI*sizeof(_Float16));
    _Float16* wouth = (_Float16*)carve((size_t)DM*DI*sizeof(_Float16));
    (void)in_sizes; (void)n_in; (void)out_size; (void)ws_size;

    const int packN = 256*DM + KD*CPAD*DI + DM*DI;   // 49152
    k_pack       <<<packN/256, 256, 0, stream>>>(in_proj_w, x_proj_w, out_pw,
                                                 winh, xpwh, wouth);
    k_ln         <<<NP/8, 256, 0, stream>>>(x, ln1_g, ln1_b, xnh);
    k_gemm_inproj<<<(NP/16)*16/4, 128, 0, stream>>>(xnh, winh, Xi, Z);
    k_conv       <<<(BN*DI*LN)/256, 256, 0, stream>>>(Xi, conv_w, conv_b, xcf, xch);
    k_gemm_xproj <<<BN*KD*(LN/16)*3/4, 128, 0, stream>>>(xch, xpwh, xdbl);
    k_scan       <<<(BN*KD*DI)/16, 256, 0, stream>>>(xdbl, xcf, dt_proj_w, dt_proj_b,
                                                     A_logs, Ds, ybuf);
    k_combine    <<<NP, 128, 0, stream>>>(ybuf, Z, out_ng, out_nb, yh16);
    k_gemm_out   <<<(NP/16)*4/4, 128, 0, stream>>>(yh16, wouth, x, out);
}
